// RandLANet_41644002902324
// MI455X (gfx1250) — compile-verified, matrix operand order checked by reading, not run on hardware
//
#include <hip/hip_runtime.h>
#include <hip/hip_bf16.h>
#include <math.h>

// ---------------------------------------------------------------------------
// RandLA-Net forward for MI455X (gfx1250, wave32).
// All GEMMs go through v_wmma_f32_16x16x32_f16 (f32 accumulate).
// K=16 neighbors == one WMMA M-tile, so attentive pooling is fused:
// gather+concat -> WMMA scores -> softmax over K -> weighted sum, per wave.
// A-tiles are staged in LDS as f16 so each WMMA A fragment is exactly two
// 16-byte ds_load_b128 per lane (ISA 7.12.2 layout: spans k=hi*8.. and 16+hi*8..).
// Workspace requirement: ~80 MB of d_ws (all intermediates fit in L2).
// Params assumed flattened in JAX pytree order (sorted dict keys) after the
// 17 array inputs.
// ---------------------------------------------------------------------------

typedef __attribute__((ext_vector_type(16))) _Float16 v16h;
typedef __attribute__((ext_vector_type(8)))  _Float16 v8h;
typedef __attribute__((ext_vector_type(4)))  _Float16 v4h;
typedef __attribute__((ext_vector_type(8)))  float    v8f;

#define BN_SCALE 0.9999995f   // 1/sqrt(1+1e-6)
#define KNBR 16

// k index inside a 16-bit 16x32 A/B WMMA fragment for (lane, half j)
__device__ __forceinline__ int frag_k(int lane, int j) {
  return ((j & 8) << 1) | ((lane >> 4) << 3) | (j & 7);
}

union frag16 { v16h v; v8h h[2]; };

// Build the A fragment for `row` from a 16x32 f16 LDS tile: two b128 DS loads.
__device__ __forceinline__ v16h load_frag_row(const _Float16* rowp, int hi) {
  frag16 u;
  u.h[0] = *(const v8h*)(rowp + hi * 8);
  u.h[1] = *(const v8h*)(rowp + 16 + hi * 8);
  return u.v;
}

// ---------------------------------------------------------------------------
// Generic fused linear: Y = act(bn(X*W + b))  (bn/act optional)
// X: (R, Din) f32 row-major, W: (Din, Dout) f32 row-major.
// One wave per 16x16 output tile; Din zero-padded to multiples of 32.
// ---------------------------------------------------------------------------
__global__ void k_linear_wmma(const float* __restrict__ X, const float* __restrict__ W,
                              const float* __restrict__ bias,
                              const float* __restrict__ gamma, const float* __restrict__ beta,
                              float* __restrict__ Y, long R, int Din, int Dout,
                              int has_bn, int has_act, float alpha)
{
  __shared__ _Float16 Ah[4][16][32] __attribute__((aligned(16)));
  const int wave = threadIdx.x >> 5;
  const int lane = threadIdx.x & 31;
  const int colTiles = (Dout + 15) >> 4;
  const long rowTiles = R >> 4;
  const long total = rowTiles * (long)colTiles;
  long tile = (long)blockIdx.x * 4 + wave;
  const bool active = tile < total;
  if (!active) tile = total - 1;        // clamp: keep barriers uniform
  const int  ct = (int)(tile % colTiles);
  const long rt = tile / colTiles;
  const long rowbase = rt << 4;
  const int  colbase = ct << 4;
  const int  m  = lane & 15;
  const int  hi = lane >> 4;
  const int  col = colbase + m;

  v8f acc = {0.f, 0.f, 0.f, 0.f, 0.f, 0.f, 0.f, 0.f};
  const int chunks = (Din + 31) >> 5;
  const bool vec4 = ((Din & 3) == 0);
  for (int c = 0; c < chunks; ++c) {
    const int kbase = c << 5;
    if (vec4 && (kbase + 32 <= Din)) {
      // branchless fast path: 4x float4 coalesced loads + packed f16 LDS stores
      #pragma unroll
      for (int t = 0; t < 4; ++t) {
        const int flat = t * 128 + lane * 4;
        const int r = flat >> 5, cc = flat & 31;
        const float4 v = *(const float4*)(X + (rowbase + r) * (long)Din + kbase + cc);
        v4h h; h[0] = (_Float16)v.x; h[1] = (_Float16)v.y;
        h[2] = (_Float16)v.z; h[3] = (_Float16)v.w;
        *(v4h*)&Ah[wave][r][cc] = h;
      }
    } else {
      #pragma unroll
      for (int i = 0; i < 16; ++i) {
        const int k = kbase + lane;
        Ah[wave][i][lane] = (_Float16)((k < Din) ? X[(rowbase + i) * (long)Din + k] : 0.f);
      }
    }
    __syncthreads();
    const v16h a = load_frag_row(&Ah[wave][m][0], hi);
    v16h bf;
    #pragma unroll
    for (int j = 0; j < 16; ++j) {
      const int kg = kbase + frag_k(lane, j);
      bf[j] = (_Float16)((kg < Din && col < Dout) ? W[(long)kg * Dout + col] : 0.f);
    }
    acc = __builtin_amdgcn_wmma_f32_16x16x32_f16(false, a, false, bf, (short)0, acc,
                                                 false, false);
    __syncthreads();
  }
  if (active && col < Dout) {
    const float bb = bias  ? bias[col]  : 0.f;
    const float g  = has_bn ? gamma[col] : 1.f;
    const float be = has_bn ? beta[col]  : 0.f;
    #pragma unroll
    for (int i = 0; i < 8; ++i) {
      float v = acc[i] + bb;
      if (has_bn) v = g * (v * BN_SCALE) + be;
      if (has_act) v = (v >= 0.f) ? v : alpha * v;
      Y[(rowbase + hi * 8 + i) * (long)Dout + col] = v;
    }
  }
}

// ---------------------------------------------------------------------------
// Fused relative-position encoding + xyz_mlp1 (10 -> dh, bn + lrelu 0.2).
// One wave per point; its K=16 neighbors form the 16-row A tile (10 ch,
// zero-padded to 32), so the rel-pos tensor is never materialized.
// ---------------------------------------------------------------------------
__global__ void k_relpos_mlp(const float* __restrict__ xyz, const int* __restrict__ neigh,
                             const float* __restrict__ W, const float* __restrict__ bias,
                             const float* __restrict__ gamma, const float* __restrict__ beta,
                             float* __restrict__ out, int n, int dh)
{
  __shared__ _Float16 A[4][16][32] __attribute__((aligned(16)));
  const int wave = threadIdx.x >> 5;
  const int lane = threadIdx.x & 31;
  const int p = blockIdx.x * 4 + wave;     // 0 .. 2n-1 (B=2)
  const int b = p / n;

  #pragma unroll
  for (int i = 0; i < 16; ++i) A[wave][i][lane] = (_Float16)0.f;  // zero pad 10 -> 32
  if (lane < 16) {
    const int k  = lane;
    const int nb = neigh[(size_t)p * KNBR + k];
    const float* c = xyz + (size_t)p * 3;
    const float* q = xyz + ((size_t)b * n + nb) * 3;
    const float rx = c[0] - q[0], ry = c[1] - q[1], rz = c[2] - q[2];
    const float dist = sqrtf(rx * rx + ry * ry + rz * rz);
    A[wave][k][0] = (_Float16)dist;
    A[wave][k][1] = (_Float16)rx;   A[wave][k][2] = (_Float16)ry;  A[wave][k][3] = (_Float16)rz;
    A[wave][k][4] = (_Float16)c[0]; A[wave][k][5] = (_Float16)c[1]; A[wave][k][6] = (_Float16)c[2];
    A[wave][k][7] = (_Float16)q[0]; A[wave][k][8] = (_Float16)q[1]; A[wave][k][9] = (_Float16)q[2];
  }
  __syncthreads();

  const int m = lane & 15, hi = lane >> 4;
  const v16h a = load_frag_row(&A[wave][m][0], hi);

  const int colTiles = (dh + 15) >> 4;
  for (int ct = 0; ct < colTiles; ++ct) {
    v16h bf;
    const int col = ct * 16 + m;
    #pragma unroll
    for (int j = 0; j < 16; ++j) {
      const int kk = frag_k(lane, j);
      bf[j] = (_Float16)((kk < 10 && col < dh) ? W[kk * dh + col] : 0.f);
    }
    v8f acc = {0.f, 0.f, 0.f, 0.f, 0.f, 0.f, 0.f, 0.f};
    acc = __builtin_amdgcn_wmma_f32_16x16x32_f16(false, a, false, bf, (short)0, acc,
                                                 false, false);
    if (col < dh) {
      const float bb = bias[col], g = gamma[col], be = beta[col];
      #pragma unroll
      for (int i = 0; i < 8; ++i) {
        float v = g * ((acc[i] + bb) * BN_SCALE) + be;
        v = (v >= 0.f) ? v : 0.2f * v;
        out[((size_t)p * KNBR + hi * 8 + i) * dh + col] = v;
      }
    }
  }
}

// ---------------------------------------------------------------------------
// Fused attentive pooling. One wave per point:
//   X = [gather(fsrc, idx) | fxyz]  (16 x d, built into LDS as f16,
//       channel-padded to a multiple of 32 so frag reads are unguarded)
//   S = X * W + b                   (WMMA, d x d)
//   P = softmax(S) over the 16 neighbors (per channel, cross-half-wave shfl)
//   agg[c] = sum_k X[k,c] * P[k,c]  -> out (P, d)
// ---------------------------------------------------------------------------
__global__ void k_attpool(const float* __restrict__ fsrc, const float* __restrict__ fxyz,
                          const int* __restrict__ neigh,
                          const float* __restrict__ W, const float* __restrict__ bias,
                          float* __restrict__ out, int n, int dh, int d)
{
  __shared__ _Float16 X[4][16][256] __attribute__((aligned(16)));
  const int wave = threadIdx.x >> 5;
  const int lane = threadIdx.x & 31;
  const int p = blockIdx.x * 4 + wave;
  const int b = p / n;
  const int dpad = (d + 31) & ~31;

  for (int k = 0; k < KNBR; ++k) {
    const int nb = neigh[(size_t)p * KNBR + k];
    const size_t src  = ((size_t)b * n + nb) * dh;
    const size_t xsrc = ((size_t)p * KNBR + k) * dh;
    for (int c = lane; c < dpad; c += 32) {
      float v = 0.f;
      if (c < dh)      v = fsrc[src + c];
      else if (c < d)  v = fxyz[xsrc + (c - dh)];
      X[wave][k][c] = (_Float16)v;
    }
  }
  __syncthreads();

  const int m = lane & 15, hi = lane >> 4;
  const int chunks = dpad >> 5;
  const int colTiles = d >> 4;
  for (int ct = 0; ct < colTiles; ++ct) {
    v8f acc = {0.f, 0.f, 0.f, 0.f, 0.f, 0.f, 0.f, 0.f};
    const int col = ct * 16 + m;
    for (int kc = 0; kc < chunks; ++kc) {
      const v16h a = load_frag_row(&X[wave][m][kc * 32], hi);
      v16h bf;
      #pragma unroll
      for (int j = 0; j < 16; ++j) {
        const int kk = kc * 32 + frag_k(lane, j);
        bf[j] = (_Float16)((kk < d) ? W[(size_t)kk * d + col] : 0.f);
      }
      acc = __builtin_amdgcn_wmma_f32_16x16x32_f16(false, a, false, bf, (short)0, acc,
                                                   false, false);
    }
    // softmax over the 16 neighbor rows for this channel (split across lane^16)
    const float bb = bias[col];
    float s[8], e[8];
    float mx = -3.0e38f;
    #pragma unroll
    for (int i = 0; i < 8; ++i) { s[i] = acc[i] + bb; mx = fmaxf(mx, s[i]); }
    mx = fmaxf(mx, __shfl_xor(mx, 16));
    float es = 0.f;
    #pragma unroll
    for (int i = 0; i < 8; ++i) { e[i] = __expf(s[i] - mx); es += e[i]; }
    es += __shfl_xor(es, 16);
    const float inv = 1.f / es;
    float ag = 0.f;
    #pragma unroll
    for (int i = 0; i < 8; ++i) {
      const int r = hi * 8 + i;
      ag += (float)X[wave][r][col] * (e[i] * inv);
    }
    ag += __shfl_xor(ag, 16);
    if (lane < 16) out[(size_t)p * d + col] = ag;
  }
}

// ---------------------------------------------------------------------------
// Elementwise / gather kernels
// ---------------------------------------------------------------------------
__global__ void k_add_lrelu(const float* __restrict__ a, const float* __restrict__ b,
                            float* __restrict__ o, size_t cnt, float alpha)
{
  size_t i = (size_t)blockIdx.x * blockDim.x + threadIdx.x;
  if (i >= cnt) return;
  float v = a[i] + b[i];
  o[i] = (v >= 0.f) ? v : alpha * v;
}

__global__ void k_maxpool(const float* __restrict__ fe, const int* __restrict__ sub,
                          float* __restrict__ out, int n_in, int n_sub, int C)
{
  size_t idx = (size_t)blockIdx.x * blockDim.x + threadIdx.x;
  size_t total = (size_t)2 * n_sub * C;      // B = 2
  if (idx >= total) return;
  size_t r = idx / C; int c = (int)(idx % C);
  int b = (int)(r / n_sub);
  const int* si = sub + r * KNBR;
  float m = -3.0e38f;
  #pragma unroll
  for (int k = 0; k < KNBR; ++k) {
    int j = si[k];
    m = fmaxf(m, fe[((size_t)b * n_in + j) * C + c]);
  }
  out[idx] = m;
}

__global__ void k_interp_concat(const float* __restrict__ enc, const float* __restrict__ f,
                                const int* __restrict__ up, float* __restrict__ cat,
                                int n_out, int n_in, int C1, int C2)
{
  const int Ct = C1 + C2;
  size_t idx = (size_t)blockIdx.x * blockDim.x + threadIdx.x;
  size_t total = (size_t)2 * n_out * Ct;     // B = 2
  if (idx >= total) return;
  size_t r = idx / Ct; int c = (int)(idx % Ct);
  int b = (int)(r / n_out);
  float v;
  if (c < C1) v = enc[r * (size_t)C1 + c];
  else {
    int s = up[r];
    v = f[((size_t)b * n_in + s) * C2 + (c - C1)];
  }
  cat[idx] = v;
}

// ---------------------------------------------------------------------------
// Host side
// ---------------------------------------------------------------------------
struct DenseP { const float *W, *b; };
struct ConvP  { const float *W, *b, *beta, *gamma; };
struct EncP {
  DenseP att1_fc; ConvP att1_mlp; DenseP att2_fc;
  ConvP att2_mlp, mlp1, res_mlp2, shortcut, xyz_mlp1, xyz_mlp2;
};

static void launch_linear(hipStream_t s, const float* X, const float* W, const float* b,
                          const float* gamma, const float* beta, float* Y,
                          long R, int Din, int Dout, int has_bn, int has_act, float alpha)
{
  long tiles = (R >> 4) * (long)((Dout + 15) >> 4);
  int blocks = (int)((tiles + 3) >> 2);
  k_linear_wmma<<<blocks, 128, 0, s>>>(X, W, b, gamma, beta, Y, R, Din, Dout,
                                       has_bn, has_act, alpha);
}

extern "C" void kernel_launch(void* const* d_in, const int* in_sizes, int n_in,
                              void* d_out, int out_size, void* d_ws, size_t ws_size,
                              hipStream_t stream)
{
  (void)in_sizes; (void)n_in; (void)out_size; (void)ws_size;
  const int B = 2, N0 = 16384;
  const int NSv[4]    = {16384, 4096, 1024, 256};
  const int NSSUBv[4] = {4096, 1024, 256, 64};
  const int Dv[4]     = {16, 64, 128, 256};

  const float* features = (const float*)d_in[0];
  const float* xyz[4]; const int *neigh[4], *sub[4], *up[4];
  for (int i = 0; i < 4; ++i) {
    xyz[i]   = (const float*)d_in[1 + 4 * i];
    neigh[i] = (const int*)  d_in[2 + 4 * i];
    sub[i]   = (const int*)  d_in[3 + 4 * i];
    up[i]    = (const int*)  d_in[4 + 4 * i];
  }

  // --- params (JAX pytree flatten order; leaf order inside dicts is sorted:
  //     'W' < 'b' < 'beta' < 'gamma') ---
  int cur = 17;
  auto nxt      = [&]() { return (const float*)d_in[cur++]; };
  auto getDense = [&]() { DenseP d; d.W = nxt(); d.b = nxt(); return d; };
  auto getConv  = [&]() { ConvP c; c.W = nxt(); c.b = nxt(); c.beta = nxt(); c.gamma = nxt(); return c; };

  const float* bn0_beta  = nxt();
  const float* bn0_gamma = nxt();
  ConvP dec_[4]; for (int j = 0; j < 4; ++j) dec_[j] = getConv();
  ConvP dec0 = getConv();
  EncP enc[4];
  for (int i = 0; i < 4; ++i) {
    enc[i].att1_fc  = getDense();
    enc[i].att1_mlp = getConv();
    enc[i].att2_fc  = getDense();
    enc[i].att2_mlp = getConv();
    enc[i].mlp1     = getConv();
    enc[i].res_mlp2 = getConv();
    enc[i].shortcut = getConv();
    enc[i].xyz_mlp1 = getConv();
    enc[i].xyz_mlp2 = getConv();
  }
  DenseP fc0 = getDense();
  ConvP  fc1 = getConv(), fc2 = getConv();
  DenseP fc3 = getDense();

  // --- workspace layout (floats), ~80 MB total ---
  float* w = (float*)d_ws;
  size_t off = 0;
  auto alloc = [&](size_t nf) { float* p = w + off; off += (nf + 63) & ~(size_t)63; return p; };
  float* F0      = alloc((size_t)B * N0 * 8);
  float* ENC_FE0 = alloc((size_t)B * N0 * 32);
  float* ENC_F[4];
  ENC_F[0] = alloc((size_t)B * 4096 * 32);
  ENC_F[1] = alloc((size_t)B * 1024 * 128);
  ENC_F[2] = alloc((size_t)B * 256 * 256);
  ENC_F[3] = alloc((size_t)B * 64 * 512);
  float* FXYZ1 = alloc(4194304);
  float* FXYZ2 = alloc(4194304);
  float* FHALF = alloc(262144);
  float* AGG1  = alloc(524288);
  float* FAGG  = alloc(262144);
  float* AGG2  = alloc(524288);
  float* BBUF  = alloc(524288);
  float* YBUF  = alloc(1048576);
  float* SCBUF = alloc(1048576);
  float* FEBUF = alloc(1048576);
  float* DECA  = alloc(1048576);
  float* DECB  = alloc(1048576);
  float* CAT   = alloc(2097152);

  // fc0 + bn0 + lrelu(0.3)
  launch_linear(stream, features, fc0.W, fc0.b, bn0_gamma, bn0_beta, F0,
                (long)B * N0, 6, 8, 1, 1, 0.3f);

  const float* f = F0; int din = 8;
  for (int i = 0; i < 4; ++i) {
    const int n = NSv[i], d = Dv[i], dh = d >> 1;
    const long P = (long)B * n, RK = P * KNBR;
    const EncP& L = enc[i];

    launch_linear(stream, f, L.mlp1.W, L.mlp1.b, L.mlp1.gamma, L.mlp1.beta,
                  FHALF, P, din, dh, 1, 1, 0.2f);
    k_relpos_mlp<<<(int)(P >> 2), 128, 0, stream>>>(
        xyz[i], neigh[i], L.xyz_mlp1.W, L.xyz_mlp1.b, L.xyz_mlp1.gamma, L.xyz_mlp1.beta,
        FXYZ1, n, dh);
    k_attpool<<<(int)(P >> 2), 128, 0, stream>>>(
        FHALF, FXYZ1, neigh[i], L.att1_fc.W, L.att1_fc.b, AGG1, n, dh, d);
    launch_linear(stream, AGG1, L.att1_mlp.W, L.att1_mlp.b, L.att1_mlp.gamma, L.att1_mlp.beta,
                  FAGG, P, d, dh, 1, 1, 0.2f);
    launch_linear(stream, FXYZ1, L.xyz_mlp2.W, L.xyz_mlp2.b, L.xyz_mlp2.gamma, L.xyz_mlp2.beta,
                  FXYZ2, RK, dh, dh, 1, 1, 0.2f);
    k_attpool<<<(int)(P >> 2), 128, 0, stream>>>(
        FAGG, FXYZ2, neigh[i], L.att2_fc.W, L.att2_fc.b, AGG2, n, dh, d);
    launch_linear(stream, AGG2, L.att2_mlp.W, L.att2_mlp.b, L.att2_mlp.gamma, L.att2_mlp.beta,
                  BBUF, P, d, d, 1, 1, 0.2f);
    launch_linear(stream, BBUF, L.res_mlp2.W, L.res_mlp2.b, L.res_mlp2.gamma, L.res_mlp2.beta,
                  YBUF, P, d, 2 * d, 1, 0, 0.2f);
    launch_linear(stream, f, L.shortcut.W, L.shortcut.b, L.shortcut.gamma, L.shortcut.beta,
                  SCBUF, P, din, 2 * d, 1, 0, 0.2f);

    float* fe = (i == 0) ? ENC_FE0 : FEBUF;
    const size_t cnt = (size_t)P * 2 * d;
    k_add_lrelu<<<(int)((cnt + 255) >> 8), 256, 0, stream>>>(YBUF, SCBUF, fe, cnt, 0.2f);
    const size_t pcnt = (size_t)B * NSSUBv[i] * 2 * d;
    k_maxpool<<<(int)((pcnt + 255) >> 8), 256, 0, stream>>>(fe, sub[i], ENC_F[i],
                                                            n, NSSUBv[i], 2 * d);
    f = ENC_F[i]; din = 2 * d;
  }

  // decoder head
  launch_linear(stream, ENC_F[3], dec0.W, dec0.b, dec0.gamma, dec0.beta, DECA,
                (long)B * 64, 512, 512, 1, 1, 0.2f);

  const float* encfeat[4] = {ENC_F[2], ENC_F[1], ENC_F[0], ENC_FE0};
  const int C1v[4]   = {256, 128, 32, 32};
  const int noutv[4] = {256, 1024, 4096, 16384};
  const int ninv[4]  = {64, 256, 1024, 4096};
  const int* upv[4]  = {up[3], up[2], up[1], up[0]};
  const float* fcur = DECA; int dfeat = 512;
  float* pp[2] = {DECB, DECA};
  for (int j = 0; j < 4; ++j) {
    const int n_out = noutv[j], n_in = ninv[j], C1 = C1v[j];
    const size_t cnt = (size_t)B * n_out * (C1 + dfeat);
    k_interp_concat<<<(int)((cnt + 255) >> 8), 256, 0, stream>>>(
        encfeat[j], fcur, upv[j], CAT, n_out, n_in, C1, dfeat);
    float* fnext = pp[j & 1];
    launch_linear(stream, CAT, dec_[j].W, dec_[j].b, dec_[j].gamma, dec_[j].beta, fnext,
                  (long)B * n_out, C1 + dfeat, C1, 1, 1, 0.2f);
    fcur = fnext; dfeat = C1;
  }

  launch_linear(stream, fcur, fc1.W, fc1.b, fc1.gamma, fc1.beta, CAT,
                (long)B * N0, 32, 64, 1, 1, 0.2f);
  launch_linear(stream, CAT, fc2.W, fc2.b, fc2.gamma, fc2.beta, DECB,
                (long)B * N0, 64, 32, 1, 1, 0.2f);
  launch_linear(stream, DECB, fc3.W, fc3.b, nullptr, nullptr, (float*)d_out,
                (long)B * N0, 32, 13, 0, 0, 0.f);
}
